// InfoNCE_20340965114462
// MI455X (gfx1250) — compile-verified, matrix operand order checked by reading, not run on hardware
//
#include <hip/hip_runtime.h>
#include <hip/hip_bf16.h>
#include <math.h>

typedef __attribute__((ext_vector_type(16))) _Float16 v16h;
typedef __attribute__((ext_vector_type(8)))  _Float16 v8h;
typedef __attribute__((ext_vector_type(8)))  float    v8f;
typedef __attribute__((ext_vector_type(4)))  int      v4i_t;

#define N2        8192      // 2N rows
#define NHALF     4096
#define DDIM      512
#define KSTEPS    16        // 512 / 32
#define TEMP_INV  10.0f     // 1 / 0.1
#define SHIFT     10.0f     // fixed logsumexp shift (cos/T <= 10 always)
#define NSPLIT    4         // column split across blockIdx.y
#define NPANEL    16        // B columns per LDS panel (double-buffered)
#define MBLOCK    128       // rows per block = 8 waves * 16

#if defined(__has_builtin)
#if __has_builtin(__builtin_amdgcn_global_load_async_to_lds_b128)
#define HAVE_ASYNC_LDS 1
#endif
#if __has_builtin(__builtin_amdgcn_s_wait_asynccnt)
#define HAVE_WAIT_ASYNC_BUILTIN 1
#endif
#endif

// pointer-to-AS-qualified-v4i types (leading attribute binds to the pointee)
typedef __attribute__((address_space(1))) v4i_t* p1_v4i;
typedef __attribute__((address_space(3))) v4i_t* p3_v4i;

__device__ __forceinline__ void wait_asynccnt_le4() {
#ifdef HAVE_ASYNC_LDS
#ifdef HAVE_WAIT_ASYNC_BUILTIN
    __builtin_amdgcn_s_wait_asynccnt(4);
#else
    asm volatile("s_wait_asynccnt 0x4" ::: "memory");
#endif
#endif
}
__device__ __forceinline__ void wait_asynccnt_le0() {
#ifdef HAVE_ASYNC_LDS
#ifdef HAVE_WAIT_ASYNC_BUILTIN
    __builtin_amdgcn_s_wait_asynccnt(0);
#else
    asm volatile("s_wait_asynccnt 0x0" ::: "memory");
#endif
#endif
}

// ---------------------------------------------------------------------------
// Kernel 1: cosine-normalize rows of [z1; z2] -> f16 matrix, zero accumulators
// ---------------------------------------------------------------------------
__global__ __launch_bounds__(256)
void normalize_f16_kernel(const float* __restrict__ z1, const float* __restrict__ z2,
                          _Float16* __restrict__ uf,
                          float* __restrict__ sumexp, float* __restrict__ posv,
                          float* __restrict__ acc) {
    const int row = blockIdx.x;
    const int tid = threadIdx.x;
    const float* src = (row < NHALF) ? (z1 + (size_t)row * DDIM)
                                     : (z2 + (size_t)(row - NHALF) * DDIM);
    float x0 = src[tid];
    float x1 = src[tid + 256];
    float ss = x0 * x0 + x1 * x1;
#pragma unroll
    for (int m = 16; m >= 1; m >>= 1) ss += __shfl_xor(ss, m, 32);

    __shared__ float red[8];
    __shared__ float stot;
    const int wave = tid >> 5, lane = tid & 31;
    if (lane == 0) red[wave] = ss;
    __syncthreads();
    if (tid == 0) {
        float t = 0.f;
#pragma unroll
        for (int i = 0; i < 8; ++i) t += red[i];
        stot = t;
        sumexp[row] = 0.f;
        posv[row]   = 0.f;
        if (row == 0) *acc = 0.f;
    }
    __syncthreads();
    const float inv = 1.0f / fmaxf(sqrtf(stot), 1e-8f);
    uf[(size_t)row * DDIM + tid]       = (_Float16)(x0 * inv);
    uf[(size_t)row * DDIM + tid + 256] = (_Float16)(x1 * inv);
}

// ---------------------------------------------------------------------------
// Kernel 2: fused Gram-matrix WMMA GEMM + masked, shifted exp accumulation.
// A strips in registers (WMMA 16-bit A layout); B panels double-buffered in
// LDS via async global->LDS DMA (ASYNCcnt) overlapped with WMMA compute.
// ---------------------------------------------------------------------------
__global__ __launch_bounds__(256)
void simclr_gemm_lse_kernel(const _Float16* __restrict__ uf,
                            float* __restrict__ sumexp,
                            float* __restrict__ posv) {
    __shared__ __align__(16) _Float16 Bsh[2][NPANEL][DDIM];   // 32 KB

    const int tid     = threadIdx.x;
    const int lane    = tid & 31;
    const int wave    = tid >> 5;
    const int halfsel = lane >> 4;       // 0: lanes 0-15, 1: lanes 16-31
    const int ko      = halfsel * 8;     // K interleave offset per ISA layout
    const int l15     = lane & 15;
    const int m0      = blockIdx.x * MBLOCK + wave * 16;

    // --- A strip (16 rows x 512) resident in registers, WMMA A layout ------
    v16h areg[KSTEPS];
    {
        const _Float16* arow = uf + (size_t)(m0 + l15) * DDIM;
#pragma unroll
        for (int ks = 0; ks < KSTEPS; ++ks) {
            v8h lo = *(const v8h*)(arow + ks * 32 + ko);
            v8h hi = *(const v8h*)(arow + ks * 32 + 16 + ko);
            areg[ks] = __builtin_shufflevector(lo, hi,
                        0,1,2,3,4,5,6,7,8,9,10,11,12,13,14,15);
        }
    }

    float se[8], pv[8];
#pragma unroll
    for (int r = 0; r < 8; ++r) { se[r] = 0.f; pv[r] = 0.f; }

    const int nbeg = blockIdx.y * (N2 / NSPLIT);
    const int P    = (N2 / NSPLIT) / NPANEL;

    // panel copy: 16 cols x 512 halves = 16 KB; 256 threads x 4 B128 each
    const int prow = tid >> 4;           // 0..15 : panel column (row of U)
    const int pcol = (tid & 15) * 32;    // 32 halves = 64 B per thread
    auto panel_load = [&](int nb, int buf) {
        const _Float16* g = uf + (size_t)(nb + prow) * DDIM + pcol;
        _Float16* s = &Bsh[buf][prow][pcol];
#ifdef HAVE_ASYNC_LDS
#pragma unroll
        for (int i = 0; i < 4; ++i)
            __builtin_amdgcn_global_load_async_to_lds_b128(
                (p1_v4i)(g + i * 8), (p3_v4i)(s + i * 8), 0, 0);
#else
#pragma unroll
        for (int i = 0; i < 4; ++i)
            *(v8h*)(s + i * 8) = *(const v8h*)(g + i * 8);
#endif
    };

    panel_load(nbeg, 0);                  // prologue: panel 0 -> buffer 0

    for (int p = 0; p < P; ++p) {
        const int nb  = nbeg + p * NPANEL;
        const int buf = p & 1;
        if (p + 1 < P) {                  // prefetch next panel into other buf
            panel_load(nb + NPANEL, buf ^ 1);
            wait_asynccnt_le4();          // panel p drained, p+1 in flight
        } else {
            wait_asynccnt_le0();
        }
        __syncthreads();                  // all waves' panel-p data visible

        const int n = nb + l15;           // this lane's output column
        v8f c = {};
        const _Float16* bbase = &Bsh[buf][l15][0];
#pragma unroll
        for (int ks = 0; ks < KSTEPS; ++ks) {
            v8h lo = *(const v8h*)(bbase + ks * 32 + ko);
            v8h hi = *(const v8h*)(bbase + ks * 32 + 16 + ko);
            v16h b = __builtin_shufflevector(lo, hi,
                        0,1,2,3,4,5,6,7,8,9,10,11,12,13,14,15);
            c = __builtin_amdgcn_wmma_f32_16x16x32_f16(
                    false, areg[ks], false, b, (short)0, c, false, false);
        }
        // C/D layout: VGPR r -> M = r + 8*halfsel, N = lane&15
#pragma unroll
        for (int r = 0; r < 8; ++r) {
            const int m     = m0 + r + halfsel * 8;
            const float s10 = c[r] * TEMP_INV;
            se[r] += (m == n) ? 0.f : __expf(s10 - SHIFT);
            if (n == ((m + NHALF) & (N2 - 1))) pv[r] += s10;
        }
        __syncthreads();                  // buf free for iteration p+1 refill
    }

    // reduce across the 16 lanes of each half-group (masks 1,2,4,8 stay in-group)
#pragma unroll
    for (int r = 0; r < 8; ++r) {
#pragma unroll
        for (int m = 8; m >= 1; m >>= 1) {
            se[r] += __shfl_xor(se[r], m, 32);
            pv[r] += __shfl_xor(pv[r], m, 32);
        }
    }
    if (l15 == 0) {
#pragma unroll
        for (int r = 0; r < 8; ++r) {
            const int m = m0 + r + halfsel * 8;
            atomicAdd(&sumexp[m], se[r]);
            atomicAdd(&posv[m],   pv[r]);
        }
    }
}

// ---------------------------------------------------------------------------
// Kernel 3: per-row nll = -pos + SHIFT + log(sumexp), block-reduce into acc
// ---------------------------------------------------------------------------
__global__ __launch_bounds__(256)
void rowreduce_kernel(const float* __restrict__ sumexp,
                      const float* __restrict__ posv,
                      float* __restrict__ acc) {
    const int r = blockIdx.x * 256 + threadIdx.x;
    float nll = -posv[r] + SHIFT + __logf(sumexp[r]);
#pragma unroll
    for (int m = 16; m >= 1; m >>= 1) nll += __shfl_xor(nll, m, 32);
    __shared__ float red[8];
    if ((threadIdx.x & 31) == 0) red[threadIdx.x >> 5] = nll;
    __syncthreads();
    if (threadIdx.x == 0) {
        float t = 0.f;
#pragma unroll
        for (int i = 0; i < 8; ++i) t += red[i];
        atomicAdd(acc, t);
    }
}

__global__ void finalize_kernel(const float* __restrict__ acc, float* __restrict__ out) {
    out[0] = acc[0] * (1.0f / (float)N2);
}

// ---------------------------------------------------------------------------
extern "C" void kernel_launch(void* const* d_in, const int* in_sizes, int n_in,
                              void* d_out, int out_size, void* d_ws, size_t ws_size,
                              hipStream_t stream) {
    (void)in_sizes; (void)n_in; (void)out_size; (void)ws_size;
    const float* z1 = (const float*)d_in[0];
    const float* z2 = (const float*)d_in[1];
    float* out = (float*)d_out;

    char* ws = (char*)d_ws;
    float*    acc     = (float*)ws;                                 // 1 float
    float*    sumexp  = (float*)(ws + 256);                         // 8192 floats
    float*    posv    = (float*)(ws + 256 + (size_t)N2 * 4);        // 8192 floats
    _Float16* uf      = (_Float16*)(ws + 256 + (size_t)2 * N2 * 4); // 8192*512 f16

    normalize_f16_kernel<<<N2, 256, 0, stream>>>(z1, z2, uf, sumexp, posv, acc);

    dim3 grid(N2 / MBLOCK, NSPLIT);
    simclr_gemm_lse_kernel<<<grid, 256, 0, stream>>>(uf, sumexp, posv);

    rowreduce_kernel<<<N2 / 256, 256, 0, stream>>>(sumexp, posv, acc);
    finalize_kernel<<<1, 1, 0, stream>>>(acc, out);
}
